// KAN_FFT_linear_59914793779340
// MI455X (gfx1250) — matrix operand order repaired and checked by reading, not compiled, not run on hardware
//
#include <hip/hip_runtime.h>

typedef __attribute__((ext_vector_type(16))) _Float16 v16h;
typedef __attribute__((ext_vector_type(8)))  _Float16 v8h;
typedef __attribute__((ext_vector_type(8)))  float    v8f;
typedef __attribute__((ext_vector_type(4)))  float    v4f;

#define IDIM  256
#define JDIM  256
#define KGRID 64
#define M_WG  128            // batch rows per WG (8 m-tiles)
#define N_WG  128            // out cols per WG (8 n-tiles)
// Wf16 (workspace) layout: fragIdx = ((trig*16 + jt)*256 + i)*2 + c
//   per frag: 32 lanes x 16 f16 (lane-contiguous 32B) -> 1KB/frag, 16 MB total
#define WF16_BYTES ((size_t)2 * 16 * 256 * 2 * 32 * 16 * 2)

static __device__ __forceinline__ v16h cat8(v8h a, v8h b) {
    return __builtin_shufflevector(a, b, 0,1,2,3,4,5,6,7,8,9,10,11,12,13,14,15);
}

// ---------------- one-shot W pre-shuffle: f32 -> f16 in B-fragment order ---
__global__ __launch_bounds__(256)
void kan_shuffle_w(const float* __restrict__ W, _Float16* __restrict__ Wf)
{
    const int gid  = blockIdx.x * 256 + threadIdx.x;   // 524288 total
    const int lane = gid & 31;
    const int frag = gid >> 5;                         // ((trig*16+jt)*256+i)*2+c
    const int c    = frag & 1;
    const int i    = (frag >> 1) & 255;
    const int jt   = (frag >> 9) & 15;
    const int trig = frag >> 13;
    const int lm = lane & 15, hi = lane >> 4;

    const float* src = W + ((size_t)(trig * JDIM + jt * 16 + lm) * IDIM + i) * KGRID
                         + c * 32 + hi * 16;
    const v4f a0 = *(const v4f*)(src);
    const v4f a1 = *(const v4f*)(src + 4);
    const v4f a2 = *(const v4f*)(src + 8);
    const v4f a3 = *(const v4f*)(src + 12);
    v8h h0, h1;
    h0[0]=(_Float16)a0[0]; h0[1]=(_Float16)a0[1]; h0[2]=(_Float16)a0[2]; h0[3]=(_Float16)a0[3];
    h0[4]=(_Float16)a1[0]; h0[5]=(_Float16)a1[1]; h0[6]=(_Float16)a1[2]; h0[7]=(_Float16)a1[3];
    h1[0]=(_Float16)a2[0]; h1[1]=(_Float16)a2[1]; h1[2]=(_Float16)a2[2]; h1[3]=(_Float16)a2[3];
    h1[4]=(_Float16)a3[0]; h1[5]=(_Float16)a3[1]; h1[6]=(_Float16)a3[2]; h1[7]=(_Float16)a3[3];
    _Float16* dst = Wf + ((size_t)frag * 32 + lane) * 16;
    *(v8h*)(dst)     = h0;
    *(v8h*)(dst + 8) = h1;
}

// ---------------- B fragment loader ---------------------------------------
template<bool PRE>
static __device__ __forceinline__ v16h loadB(const _Float16* __restrict__ Wf,
                                             const float* __restrict__ W,
                                             int lane, int lm, int hi,
                                             int jt, int i, int cc)
{
    const int trig = cc >> 1, c = cc & 1;
    if constexpr (PRE) {
        const size_t frag = ((size_t)(trig * 16 + jt) * 256 + i) * 2 + c;
        const _Float16* p = Wf + (frag * 32 + lane) * 16;
        const v8h b0 = *(const v8h*)(p);
        const v8h b1 = *(const v8h*)(p + 8);
        return cat8(b0, b1);
    } else {
        const float* p = W + ((size_t)(trig * JDIM + jt * 16 + lm) * IDIM + i) * KGRID
                           + c * 32 + hi * 16;
        const v4f a0 = *(const v4f*)(p);
        const v4f a1 = *(const v4f*)(p + 4);
        const v4f a2 = *(const v4f*)(p + 8);
        const v4f a3 = *(const v4f*)(p + 12);
        v8h h0, h1;
        h0[0]=(_Float16)a0[0]; h0[1]=(_Float16)a0[1]; h0[2]=(_Float16)a0[2]; h0[3]=(_Float16)a0[3];
        h0[4]=(_Float16)a1[0]; h0[5]=(_Float16)a1[1]; h0[6]=(_Float16)a1[2]; h0[7]=(_Float16)a1[3];
        h1[0]=(_Float16)a2[0]; h1[1]=(_Float16)a2[1]; h1[2]=(_Float16)a2[2]; h1[3]=(_Float16)a2[3];
        h1[4]=(_Float16)a3[0]; h1[5]=(_Float16)a3[1]; h1[6]=(_Float16)a3[2]; h1[7]=(_Float16)a3[3];
        return cat8(h0, h1);
    }
}

// ---------------- fused main kernel ----------------------------------------
template<bool PRE>
__global__ __launch_bounds__(256)
void kan_fft_main(const float* __restrict__ X,
                  const float* __restrict__ W,
                  const _Float16* __restrict__ Wf,
                  const float* __restrict__ bias,
                  float* __restrict__ Y)
{
    // cos/sin tiles stored in exact A-fragment order:
    // frag (trig, mt 0..7, c 0..1): 32 lanes x 16 f16, 1KB each -> 32 KB
    __shared__ __align__(16) _Float16 cs[32 * 512];

    const int tid  = threadIdx.x;
    const int wave = tid >> 5;
    const int lane = tid & 31;
    const int hi = lane >> 4, lm = lane & 15;
    const int mg = wave >> 1;                 // 0..3 (m-group, 2 m-tiles each)
    const int ng = wave & 1;                  // 0..1 (n-group, 4 n-tiles each)
    const int row0   = blockIdx.x * M_WG;
    const int jtile0 = blockIdx.y * 8 + ng * 4;   // absolute 16-wide j-tile base

    // producer role: 2 threads per batch row, split k-halves
    const int prow = tid & 127;               // row within WG tile
    const int h    = tid >> 7;                // 0: k=1..32, 1: k=33..64 (wave-uniform)
    const int mt_p = prow >> 4;
    const int slot0 = (prow & 15);            // lane slot hi=0
    const int slot1 = 16 + (prow & 15);       // lane slot hi=1
    const float* xptr = X + (size_t)(row0 + prow) * IDIM;

    v8f acc[2][4];
    #pragma unroll
    for (int s = 0; s < 2; ++s)
        #pragma unroll
        for (int n = 0; n < 4; ++n) { v8f z = {}; acc[s][n] = z; }

    // prologue: prefetch B fragments for (i=0, chunk=0)
    v16h Bbuf[4];
    #pragma unroll
    for (int n = 0; n < 4; ++n)
        Bbuf[n] = loadB<PRE>(Wf, W, lane, lm, hi, jtile0 + n, 0, 0);

    #pragma unroll 1
    for (int i = 0; i < IDIM; ++i) {
        // warm WGP$ with next i's B fragments (one full phase of distance)
        if constexpr (PRE) {
            if (i + 1 < IDIM) {
                #pragma unroll
                for (int cc2 = 0; cc2 < 4; ++cc2)
                    #pragma unroll
                    for (int n = 0; n < 4; ++n) {
                        const size_t frag =
                            ((size_t)((cc2 >> 1) * 16 + jtile0 + n) * 256 + (i + 1)) * 2
                            + (cc2 & 1);
                        __builtin_prefetch(Wf + (frag * 32 + lane) * 16, 0, 1);
                    }
            }
        }

        // ---- producer: two interleaved rotation chains (step = 2x) --------
        const float xv = xptr[i];
        float s1, c1;
        __sincosf(xv, &s1, &c1);                       // 1x rotation
        float ca, sa;
        if (h) { __sincosf(33.0f * xv, &sa, &ca); }    // start at k=33
        else   { ca = c1; sa = s1; }                   // start at k=1
        float cb = __builtin_fmaf(ca, c1, -(sa * s1)); // start+1
        float sb = __builtin_fmaf(sa, c1,  (ca * s1));
        const float c2 = __builtin_fmaf(-2.0f * s1, s1, 1.0f);  // cos 2x
        const float s2 = 2.0f * c1 * s1;                        // sin 2x
        _Float16 ch[32], sh[32];
        #pragma unroll
        for (int lk = 0; lk < 32; lk += 2) {
            ch[lk]     = (_Float16)ca;  sh[lk]     = (_Float16)sa;
            ch[lk + 1] = (_Float16)cb;  sh[lk + 1] = (_Float16)sb;
            const float can = __builtin_fmaf(ca, c2, -(sa * s2));
            const float san = __builtin_fmaf(sa, c2,  (ca * s2));
            const float cbn = __builtin_fmaf(cb, c2, -(sb * s2));
            const float sbn = __builtin_fmaf(sb, c2,  (cb * s2));
            ca = can; sa = san; cb = cbn; sb = sbn;
        }
        {   // cos -> trig 0 frag, sin -> trig 1 frag (frag stride = 512 f16)
            _Float16* b0 = &cs[(size_t)(((0 * 8 + mt_p) * 2 + h)) * 512];
            _Float16* b1 = &cs[(size_t)(((1 * 8 + mt_p) * 2 + h)) * 512];
            v8h t0, t1;
            #pragma unroll
            for (int e = 0; e < 8; ++e) { t0[e] = ch[e];      t1[e] = ch[8 + e]; }
            *(v8h*)(b0 + slot0 * 16) = t0;  *(v8h*)(b0 + slot0 * 16 + 8) = t1;
            #pragma unroll
            for (int e = 0; e < 8; ++e) { t0[e] = ch[16 + e]; t1[e] = ch[24 + e]; }
            *(v8h*)(b0 + slot1 * 16) = t0;  *(v8h*)(b0 + slot1 * 16 + 8) = t1;
            #pragma unroll
            for (int e = 0; e < 8; ++e) { t0[e] = sh[e];      t1[e] = sh[8 + e]; }
            *(v8h*)(b1 + slot0 * 16) = t0;  *(v8h*)(b1 + slot0 * 16 + 8) = t1;
            #pragma unroll
            for (int e = 0; e < 8; ++e) { t0[e] = sh[16 + e]; t1[e] = sh[24 + e]; }
            *(v8h*)(b1 + slot1 * 16) = t0;  *(v8h*)(b1 + slot1 * 16 + 8) = t1;
        }
        __syncthreads();

        // ---- bulk-load all 8 A fragments for this i ------------------------
        v16h A[4][2];
        #pragma unroll
        for (int cc = 0; cc < 4; ++cc) {
            const int trig = cc >> 1, c = cc & 1;
            #pragma unroll
            for (int s = 0; s < 2; ++s) {
                const _Float16* p =
                    &cs[(size_t)(((trig * 8 + (mg * 2 + s)) * 2 + c)) * 512 + lane * 16];
                const v8h a0 = *(const v8h*)(p);
                const v8h a1 = *(const v8h*)(p + 8);
                A[cc][s] = cat8(a0, a1);
            }
        }
        __builtin_amdgcn_sched_barrier(0);   // pin A loads before WMMA phase

        // ---- WMMA phase: 4 chunks, depth-1 register-pipelined B -----------
        #pragma unroll
        for (int cc = 0; cc < 4; ++cc) {
            // prefetch next chunk's B (global loads legally cross the barrier)
            const int inext  = (cc == 3) ? ((i == IDIM - 1) ? i : i + 1) : i;
            const int ccnext = (cc + 1) & 3;
            v16h Bn[4];
            #pragma unroll
            for (int n = 0; n < 4; ++n)
                Bn[n] = loadB<PRE>(Wf, W, lane, lm, hi, jtile0 + n, inext, ccnext);
            __builtin_amdgcn_sched_barrier(0);   // loads stay above the WMMAs

            #pragma unroll
            for (int n = 0; n < 4; ++n)
                #pragma unroll
                for (int s = 0; s < 2; ++s)
                    acc[s][n] = __builtin_amdgcn_wmma_f32_16x16x32_f16(
                        false, A[cc][s], false, Bbuf[n], (short)0, acc[s][n], false, false);

            #pragma unroll
            for (int n = 0; n < 4; ++n) Bbuf[n] = Bn[n];
        }
        __syncthreads();
    }

    // ---- epilogue: bias + coalesced stores (D: VGPR r, lane half -> row) ---
    #pragma unroll
    for (int n = 0; n < 4; ++n) {
        const int col = (jtile0 + n) * 16 + lm;
        const float bv = bias[col];
        #pragma unroll
        for (int s = 0; s < 2; ++s) {
            const int rbase = row0 + (mg * 2 + s) * 16 + hi * 8;
            #pragma unroll
            for (int r = 0; r < 8; ++r)
                Y[(size_t)(rbase + r) * JDIM + col] = acc[s][n][r] + bv;
        }
    }
}

extern "C" void kernel_launch(void* const* d_in, const int* in_sizes, int n_in,
                              void* d_out, int out_size, void* d_ws, size_t ws_size,
                              hipStream_t stream)
{
    const float* X    = (const float*)d_in[0];   // (B, 256) f32
    const float* W    = (const float*)d_in[1];   // (2, 256, 256, 64) f32
    const float* bias = (const float*)d_in[2];   // (1, 256) f32
    float* Y = (float*)d_out;                    // (B, 256) f32

    const int B = in_sizes[0] / IDIM;            // 8192
    dim3 grid(B / M_WG, JDIM / N_WG, 1);         // (64, 2)
    dim3 block(256, 1, 1);

    if (ws_size >= WF16_BYTES) {
        _Float16* Wf = (_Float16*)d_ws;
        hipLaunchKernelGGL(kan_shuffle_w, dim3(2048), block, 0, stream, W, Wf);
        hipLaunchKernelGGL((kan_fft_main<true>), grid, block, 0, stream,
                           X, W, (const _Float16*)Wf, bias, Y);
    } else {
        hipLaunchKernelGGL((kan_fft_main<false>), grid, block, 0, stream,
                           X, W, (const _Float16*)nullptr, bias, Y);
    }
}
